// HGNNConv_17901423690226
// MI455X (gfx1250) — matrix operand order repaired and checked by reading, not compiled, not run on hardware
//
#include <hip/hip_runtime.h>
#include <stdint.h>

typedef __attribute__((ext_vector_type(16))) __bf16        v16bf;
typedef __attribute__((ext_vector_type(8)))  float         v8f;
typedef __attribute__((ext_vector_type(8)))  unsigned int  v8u;

#define N_NODES 50000
#define N_EDGES 25000
#define NNZ     800000
#define D       1024

// ---------- helpers ----------

union BFPack { v8u u; v16bf b; };

__device__ __forceinline__ v16bf pack16(uint4 g0, uint4 g1) {
    BFPack p;
    p.u[0] = g0.x; p.u[1] = g0.y; p.u[2] = g0.z; p.u[3] = g0.w;
    p.u[4] = g1.x; p.u[5] = g1.y; p.u[6] = g1.z; p.u[7] = g1.w;
    return p.b;
}

// Truncating fp32 -> bf16 hi/lo split: x ~= hi + lo, lo = x - float(hi) exactly.
__device__ __forceinline__ void split_bf16(float x, unsigned short& hi, unsigned short& lo) {
    unsigned int xb = __float_as_uint(x);
    hi = (unsigned short)(xb >> 16);
    float hif = __uint_as_float(xb & 0xFFFF0000u);
    float lof = x - hif;
    lo = (unsigned short)(__float_as_uint(lof) >> 16);
}

// ---------- kernel 1: degree histograms ----------
__global__ void deg_kernel(const int* __restrict__ node_idx, const int* __restrict__ edge_idx,
                           float* __restrict__ dv, float* __restrict__ de) {
    int i = blockIdx.x * blockDim.x + threadIdx.x;
    if (i < NNZ) {
        atomicAdd(&dv[node_idx[i]], 1.0f);
        atomicAdd(&de[edge_idx[i]], 1.0f);
    }
}

// mode 0: d <- rsqrt(d) if d>0 else 0 ; mode 1: d <- 1/d if d>0 else 0
__global__ void finalize_deg(float* __restrict__ d, int n, int mode) {
    int i = blockIdx.x * blockDim.x + threadIdx.x;
    if (i < n) {
        float v = d[i];
        d[i] = (v > 0.0f) ? (mode ? (1.0f / v) : rsqrtf(v)) : 0.0f;
    }
}

// ---------- kernel 2: W [K][N] fp32 -> WT [N][K] bf16 hi/lo ----------
__global__ void convw_kernel(const float* __restrict__ W,
                             unsigned short* __restrict__ WThi,
                             unsigned short* __restrict__ WTlo) {
    int idx = blockIdx.x * blockDim.x + threadIdx.x;
    if (idx < D * D) {
        int k = idx >> 10;
        int n = idx & (D - 1);
        unsigned short h, l;
        split_bf16(W[idx], h, l);
        WThi[(size_t)n * D + k] = h;
        WTlo[(size_t)n * D + k] = l;
    }
}

// ---------- kernel 3: WMMA GEMM  Xs = (X*W + b) * dv_isqrt[m] ----------
// block: 256 thr = 8 waves (4 M-waves x 2 N-waves); block tile 128M x 128N; K-step 32.
#define LDS_STRIDE 40   // ushorts per LDS row: 32 data + 8 pad (80B -> 16B aligned, conflict-free)

__global__ __launch_bounds__(256) void gemm_kernel(
    const float* __restrict__ X,
    const unsigned short* __restrict__ WThi,
    const unsigned short* __restrict__ WTlo,
    const float* __restrict__ bias,
    const float* __restrict__ dvis,
    float* __restrict__ Xs)
{
    __shared__ __align__(16) unsigned short Ahi[128 * LDS_STRIDE];
    __shared__ __align__(16) unsigned short Alo[128 * LDS_STRIDE];

    const int tid   = threadIdx.x;
    const int lane  = tid & 31;
    const int wave  = tid >> 5;
    const int waveM = wave & 3;    // 0..3 -> 32 rows each
    const int waveN = wave >> 2;   // 0..1 -> 64 cols each
    const int m0 = blockIdx.x * 128;
    const int n0 = blockIdx.y * 128 + waveN * 64;

    const int lm      = lane & 15;
    const int halfsel = lane >> 4;  // 0: K lower half, 1: K upper half (contiguous K order used by this code)

    v8f acc[2][4];
    #pragma unroll
    for (int mt = 0; mt < 2; ++mt)
        #pragma unroll
        for (int nt = 0; nt < 4; ++nt)
            #pragma unroll
            for (int r = 0; r < 8; ++r)
                acc[mt][nt][r] = 0.0f;

    for (int kt = 0; kt < D; kt += 32) {
        __syncthreads();
        // stage A tile 128x32 fp32 -> bf16 hi/lo in LDS (1024 float4s / 256 threads)
        #pragma unroll
        for (int it = 0; it < 4; ++it) {
            int idx = tid + it * 256;        // 0..1023
            int row = idx >> 3;              // 0..127
            int col = (idx & 7) << 2;        // 0,4,...,28
            int m = m0 + row;
            float4 v = make_float4(0.0f, 0.0f, 0.0f, 0.0f);
            if (m < N_NODES) {
                v = *(const float4*)(X + (size_t)m * D + kt + col);
                if (kt + 32 < D)   // prefetch next K tile (global_prefetch_b8)
                    __builtin_prefetch(X + (size_t)m * D + kt + 32 + col, 0, 3);
            }
            unsigned short h0, l0, h1, l1, h2, l2, h3, l3;
            split_bf16(v.x, h0, l0); split_bf16(v.y, h1, l1);
            split_bf16(v.z, h2, l2); split_bf16(v.w, h3, l3);
            unsigned int* ph = (unsigned int*)(Ahi + row * LDS_STRIDE + col);
            ph[0] = (unsigned int)h0 | ((unsigned int)h1 << 16);
            ph[1] = (unsigned int)h2 | ((unsigned int)h3 << 16);
            unsigned int* pl = (unsigned int*)(Alo + row * LDS_STRIDE + col);
            pl[0] = (unsigned int)l0 | ((unsigned int)l1 << 16);
            pl[1] = (unsigned int)l2 | ((unsigned int)l3 << 16);
        }
        __syncthreads();

        // B fragments straight from global (WT is [N][K], hot in L2):
        // lane holds column n = tile + lm, K = halfsel*16 .. +15 contiguous (32B).
        v16bf bh[4], bl[4];
        #pragma unroll
        for (int nt = 0; nt < 4; ++nt) {
            size_t off = (size_t)(n0 + nt * 16 + lm) * D + (kt + halfsel * 16);
            const uint4* ph = (const uint4*)(WThi + off);
            const uint4* pl = (const uint4*)(WTlo + off);
            bh[nt] = pack16(ph[0], ph[1]);
            bl[nt] = pack16(pl[0], pl[1]);
        }

        // A fragments from LDS: lane holds row m = tile + lm;
        // words {q..q+7} with q = halfsel*8 (K = halfsel*16 .. +15 contiguous)  ->  two b128 loads.
        v16bf ah[2], al[2];
        #pragma unroll
        for (int mt = 0; mt < 2; ++mt) {
            int row = waveM * 32 + mt * 16 + lm;
            const char* rh = (const char*)Ahi + row * (LDS_STRIDE * 2) + halfsel * 32;
            ah[mt] = pack16(*(const uint4*)(rh), *(const uint4*)(rh + 16));
            const char* rl = (const char*)Alo + row * (LDS_STRIDE * 2) + halfsel * 32;
            al[mt] = pack16(*(const uint4*)(rl), *(const uint4*)(rl + 16));
        }

        // bf16x3 fp32 emulation: hi*hi + hi*lo + lo*hi
        #pragma unroll
        for (int mt = 0; mt < 2; ++mt) {
            #pragma unroll
            for (int nt = 0; nt < 4; ++nt) {
                acc[mt][nt] = __builtin_amdgcn_wmma_f32_16x16x32_bf16(
                    false, ah[mt], false, bh[nt], (short)0, acc[mt][nt], false, false);
                acc[mt][nt] = __builtin_amdgcn_wmma_f32_16x16x32_bf16(
                    false, ah[mt], false, bl[nt], (short)0, acc[mt][nt], false, false);
                acc[mt][nt] = __builtin_amdgcn_wmma_f32_16x16x32_bf16(
                    false, al[mt], false, bh[nt], (short)0, acc[mt][nt], false, false);
            }
        }
    }

    // epilogue: +bias, * dv^{-1/2}; C/D layout: VGPR r -> row r (lanes 0-15) / 8+r (lanes 16-31)
    #pragma unroll
    for (int mt = 0; mt < 2; ++mt) {
        #pragma unroll
        for (int nt = 0; nt < 4; ++nt) {
            int n = n0 + nt * 16 + lm;
            float bn = bias[n];
            #pragma unroll
            for (int r = 0; r < 8; ++r) {
                int m = m0 + waveM * 32 + mt * 16 + r + halfsel * 8;
                if (m < N_NODES)
                    Xs[(size_t)m * D + n] = (acc[mt][nt][r] + bn) * dvis[m];
            }
        }
    }
}

// ---------- kernel 4: E[edge] += Xs[node]  (one block per nnz) ----------
__global__ __launch_bounds__(256) void scatter_edge_kernel(
    const float* __restrict__ Xs, const int* __restrict__ node_idx,
    const int* __restrict__ edge_idx, float* __restrict__ E)
{
    int i = blockIdx.x;
    int v = node_idx[i];
    int e = edge_idx[i];
    int t = threadIdx.x;
    float4 val = *(const float4*)(Xs + (size_t)v * D + t * 4);
    float* dst = E + (size_t)e * D + t * 4;
    atomicAdd(dst + 0, val.x);
    atomicAdd(dst + 1, val.y);
    atomicAdd(dst + 2, val.z);
    atomicAdd(dst + 3, val.w);
}

// ---------- kernel 5: out[node] += E[edge] * de_inv[edge] ----------
__global__ __launch_bounds__(256) void scatter_node_kernel(
    const float* __restrict__ E, const float* __restrict__ dei,
    const int* __restrict__ node_idx, const int* __restrict__ edge_idx,
    float* __restrict__ out)
{
    int i = blockIdx.x;
    int v = node_idx[i];
    int e = edge_idx[i];
    float s = dei[e];
    int t = threadIdx.x;
    float4 val = *(const float4*)(E + (size_t)e * D + t * 4);
    float* dst = out + (size_t)v * D + t * 4;
    atomicAdd(dst + 0, val.x * s);
    atomicAdd(dst + 1, val.y * s);
    atomicAdd(dst + 2, val.z * s);
    atomicAdd(dst + 3, val.w * s);
}

// ---------- kernel 6: out = relu(out * dv_isqrt[row]) in place ----------
__global__ void relu_scale_kernel(float* __restrict__ out, const float* __restrict__ dvis) {
    size_t i = (size_t)blockIdx.x * blockDim.x + threadIdx.x;   // over N_NODES*D/4 float4s
    if (i < (size_t)N_NODES * D / 4) {
        int m = (int)(i >> 8);      // (i*4)/1024
        float s = dvis[m];
        float4* p = (float4*)out + i;
        float4 v = *p;
        v.x = fmaxf(v.x * s, 0.0f);
        v.y = fmaxf(v.y * s, 0.0f);
        v.z = fmaxf(v.z * s, 0.0f);
        v.w = fmaxf(v.w * s, 0.0f);
        *p = v;
    }
}

// ---------- launcher ----------
extern "C" void kernel_launch(void* const* d_in, const int* in_sizes, int n_in,
                              void* d_out, int out_size, void* d_ws, size_t ws_size,
                              hipStream_t stream) {
    (void)in_sizes; (void)n_in; (void)out_size; (void)ws_size;

    const float* X        = (const float*)d_in[0];
    const int*   node_idx = (const int*)d_in[1];
    const int*   edge_idx = (const int*)d_in[2];
    const float* W        = (const float*)d_in[3];
    const float* b        = (const float*)d_in[4];
    float*       out      = (float*)d_out;

    char*  ws = (char*)d_ws;
    size_t o  = 0;
    auto alloc = [&](size_t bytes) -> char* {
        char* p = ws + o;
        o = (o + bytes + 255) & ~(size_t)255;
        return p;
    };
    float*          dv   = (float*)alloc((size_t)N_NODES * 4);
    float*          de   = (float*)alloc((size_t)N_EDGES * 4);
    unsigned short* WThi = (unsigned short*)alloc((size_t)D * D * 2);
    unsigned short* WTlo = (unsigned short*)alloc((size_t)D * D * 2);
    float*          Xs   = (float*)alloc((size_t)N_NODES * D * 4);
    float*          E    = (float*)alloc((size_t)N_EDGES * D * 4);

    // zero accumulators (graph-capture safe)
    hipMemsetAsync(dv, 0, (size_t)N_NODES * 4, stream);
    hipMemsetAsync(de, 0, (size_t)N_EDGES * 4, stream);
    hipMemsetAsync(E, 0, (size_t)N_EDGES * D * 4, stream);
    hipMemsetAsync(out, 0, (size_t)N_NODES * D * 4, stream);

    deg_kernel<<<(NNZ + 255) / 256, 256, 0, stream>>>(node_idx, edge_idx, dv, de);
    finalize_deg<<<(N_NODES + 255) / 256, 256, 0, stream>>>(dv, N_NODES, 0);
    finalize_deg<<<(N_EDGES + 255) / 256, 256, 0, stream>>>(de, N_EDGES, 1);
    convw_kernel<<<(D * D + 255) / 256, 256, 0, stream>>>(W, WThi, WTlo);

    dim3 ggrid((N_NODES + 127) / 128, D / 128);
    gemm_kernel<<<ggrid, 256, 0, stream>>>(X, WThi, WTlo, b, dv, Xs);

    scatter_edge_kernel<<<NNZ, 256, 0, stream>>>(Xs, node_idx, edge_idx, E);
    scatter_node_kernel<<<NNZ, 256, 0, stream>>>(E, de, node_idx, edge_idx, out);

    relu_scale_kernel<<<((size_t)N_NODES * D / 4 + 255) / 256, 256, 0, stream>>>(out, dv);
}
